// MolNet_MS_7275674599519
// MI455X (gfx1250) — matrix operand order repaired and checked by reading, not compiled, not run on hardware
//
#include <hip/hip_runtime.h>
#include <hip/hip_bf16.h>
#include <math.h>

// ---------------- problem constants ----------------
#define Bsz   64
#define Npts  384
#define Knn   8
#define EMB   960
#define BN    (Bsz*Npts)          // 24576
#define BNK   (Bsz*Npts*Knn)     // 196608

typedef __attribute__((ext_vector_type(16))) _Float16 v16h;
typedef __attribute__((ext_vector_type(8)))  _Float16 v8h;
typedef __attribute__((ext_vector_type(8)))  float    v8f;

// ---------------- WMMA helpers (wave32, gfx1250) ----------------
__device__ __forceinline__ v8f wmma_f16(v16h a, v16h b, v8f c) {
  // D = A(16x32 f16) * B(32x16 f16) + C(16x16 f32)
  return __builtin_amdgcn_wmma_f32_16x16x32_f16(false, a, false, b, (short)0, c,
                                                false, false);
}

// A fragment: 16x32 tile of row-major M x K (lda = padded K, multiple of 32).
// Per ISA layout: lane L holds row M=L&15; lanes<16 hold K[0..7]+K[16..23],
// lanes>=16 hold K[8..15]+K[24..31].
__device__ __forceinline__ v16h load_a_f16(const _Float16* A, int m0, int lda, int k0) {
  int lane = threadIdx.x & 31;
  int half = lane >> 4;
  const _Float16* p = A + (size_t)(m0 + (lane & 15)) * lda + k0;
  v8h lo = *(const v8h*)(p + half * 8);        // K = 8*half .. 8*half+7   -> v0..v3
  v8h hi = *(const v8h*)(p + 16 + half * 8);   // K = 16+8*half .. +7      -> v4..v7
  v16h out;
#pragma unroll
  for (int i = 0; i < 8; ++i) { out[i] = lo[i]; out[8 + i] = hi[i]; }
  return out;
}

// B fragment from a row-major "N x K" array (i.e. B^T, e.g. weights W[o][c]).
// Lane L holds column N = n0 + (L&15); lanes<16 hold K[0..7]+K[16..23], lanes>=16 hold K[8..15]+K[24..31].
__device__ __forceinline__ v16h load_bT_f16(const _Float16* Wt, int n0, int ldw, int k0) {
  int lane = threadIdx.x & 31;
  int half = lane >> 4;
  const _Float16* p = Wt + (size_t)(n0 + (lane & 15)) * ldw + k0;
  v8h lo = *(const v8h*)(p + half * 8);  // 16B aligned: ldw, k0 multiples of 32 halves
  v8h hi = *(const v8h*)(p + 16 + half * 8);
  v16h out;
#pragma unroll
  for (int i = 0; i < 8; ++i) { out[i] = lo[i]; out[8 + i] = hi[i]; }
  return out;
}

__device__ __forceinline__ v8f vzero8() {
  v8f z = {0.f,0.f,0.f,0.f,0.f,0.f,0.f,0.f};
  return z;
}

__device__ __forceinline__ float leaky(float v, float s) { return v >= 0.f ? v : s * v; }

// ---------------- prep kernels ----------------
// x (B,21,N) f32  ->  xt (B,N,32) f16, zero padded
__global__ void k_prep0(const float* __restrict__ x, _Float16* __restrict__ xt) {
  int i = blockIdx.x * blockDim.x + threadIdx.x;
  if (i >= BN) return;
  int b = i / Npts, n = i % Npts;
  for (int c = 0; c < 32; ++c) {
    float v = (c < 21) ? x[((size_t)b * 21 + c) * Npts + n] : 0.f;
    xt[(size_t)i * 32 + c] = (_Float16)v;
  }
}

// xx[b,n] = sum_c xt^2
__global__ void k_rowsq(const _Float16* __restrict__ xt, int Cp, float* __restrict__ xx) {
  int i = blockIdx.x * blockDim.x + threadIdx.x;
  if (i >= BN) return;
  const _Float16* r = xt + (size_t)i * Cp;
  float s = 0.f;
  for (int c = 0; c < Cp; ++c) { float v = (float)r[c]; s += v * v; }
  xx[i] = s;
}

// cent16 (layer 0): copy channels [3..20] of xt, zero pad to 32
__global__ void k_centcopy(const _Float16* __restrict__ xt, _Float16* __restrict__ cent,
                           int Cp, int chOff, int eff, int effp) {
  long long total = (long long)BN * effp;
  for (long long i = (long long)blockIdx.x * blockDim.x + threadIdx.x; i < total;
       i += (long long)gridDim.x * blockDim.x) {
    int c = (int)(i % effp); long long r = i / effp;
    _Float16 v = (_Float16)0.f;
    if (c < eff) v = xt[(size_t)r * Cp + chOff + c];
    cent[i] = v;
  }
}

// f32 weight (rows x cols starting at coloff, row stride ldsrc) -> f16 rows x Kp, zero pad
__global__ void k_cvt_w(const float* __restrict__ src, _Float16* __restrict__ dst,
                        int rows, int cols, int ldsrc, int coloff, int Kp) {
  long long total = (long long)rows * Kp;
  for (long long i = (long long)blockIdx.x * blockDim.x + threadIdx.x; i < total;
       i += (long long)gridDim.x * blockDim.x) {
    int c = (int)(i % Kp); int r = (int)(i / Kp);
    dst[i] = (c < cols) ? (_Float16)src[(size_t)r * ldsrc + coloff + c] : (_Float16)0.f;
  }
}

// ---------------- Gram: pair = 2*xt.xt^T - xx_n - xx_m  (2x2 register-blocked) -----------
__global__ void k_gram(const _Float16* __restrict__ xt, const float* __restrict__ xx,
                       float* __restrict__ pair, int Cp) {
  int b = blockIdx.z;
  int n0 = blockIdx.x * 32, m0 = blockIdx.y * 32;
  const _Float16* base = xt + (size_t)b * Npts * Cp;
  v8f acc[2][2];
#pragma unroll
  for (int i = 0; i < 2; ++i)
#pragma unroll
    for (int j = 0; j < 2; ++j) acc[i][j] = vzero8();
  for (int k0 = 0; k0 < Cp; k0 += 32) {
    v16h a0 = load_a_f16(base, n0,      Cp, k0);
    v16h a1 = load_a_f16(base, n0 + 16, Cp, k0);
    v16h b0 = load_bT_f16(base, m0,      Cp, k0);
    v16h b1 = load_bT_f16(base, m0 + 16, Cp, k0);
    acc[0][0] = wmma_f16(a0, b0, acc[0][0]);
    acc[0][1] = wmma_f16(a0, b1, acc[0][1]);
    acc[1][0] = wmma_f16(a1, b0, acc[1][0]);
    acc[1][1] = wmma_f16(a1, b1, acc[1][1]);
  }
  int lane = threadIdx.x & 31, col = lane & 15, half = lane >> 4;
  const float* xxb = xx + (size_t)b * Npts;
  float* pr = pair + (size_t)b * Npts * Npts;
#pragma unroll
  for (int i = 0; i < 2; ++i) {
#pragma unroll
    for (int j = 0; j < 2; ++j) {
      float xm = xxb[m0 + 16 * j + col];
#pragma unroll
      for (int q = 0; q < 8; ++q) {
        int rr = n0 + 16 * i + q + 8 * half;
        pr[(size_t)rr * Npts + m0 + 16 * j + col] =
            2.f * acc[i][j][q] - xxb[rr] - xm;
      }
    }
  }
}

// ---------------- top-K (K=8) per row of pair ----------------
__global__ void k_topk(const float* __restrict__ pair, int* __restrict__ idx,
                       float* __restrict__ dist) {
  int wave = (blockIdx.x * blockDim.x + threadIdx.x) >> 5;  // = b*N+n
  int lane = threadIdx.x & 31;
  const float* row = pair + (size_t)wave * Npts;
  float v[12];
#pragma unroll
  for (int i = 0; i < 12; ++i) v[i] = row[lane + 32 * i];
  for (int k = 0; k < Knn; ++k) {
    float bv = -3.4e38f; int bidx = 0;
#pragma unroll
    for (int i = 0; i < 12; ++i)
      if (v[i] > bv) { bv = v[i]; bidx = lane + 32 * i; }
#pragma unroll
    for (int off = 16; off; off >>= 1) {
      float ov = __shfl_xor(bv, off, 32);
      int   oi = __shfl_xor(bidx, off, 32);
      if (ov > bv || (ov == bv && oi < bidx)) { bv = ov; bidx = oi; }
    }
    if (lane == 0) {
      idx[(size_t)wave * Knn + k]  = bidx;
      dist[(size_t)wave * Knn + k] = sqrtf(fmaxf(-bv, 1e-12f));
    }
    if ((bidx & 31) == lane) v[bidx >> 5] = -3.4e38f;
  }
}

// ---------------- gather neighbors: neigh (B,N,K,effp) f16 ----------------
__global__ void k_gather(const _Float16* __restrict__ xt, const int* __restrict__ idx,
                         _Float16* __restrict__ neigh, int Cp, int eff, int effp,
                         int chOff, long long total) {
  for (long long i = (long long)blockIdx.x * blockDim.x + threadIdx.x; i < total;
       i += (long long)gridDim.x * blockDim.x) {
    int c = (int)(i % effp);
    long long rk = i / effp;               // (b*N+n)*K + k
    long long bn = rk / Knn;
    int b = (int)(bn / Npts);
    _Float16 v = (_Float16)0.f;
    if (c < eff) {
      int m = idx[rk];
      v = xt[((size_t)b * Npts + m) * Cp + chOff + c];
    }
    neigh[i] = v;
  }
}

// ------- generic WMMA GEMM, 4Mx2O register-blocked: D(MxO)=A(MxKp)*W(OxKp)^T (+bias) -----
// grid: (M/64, O/32); one wave per 64x32 output tile -> 8 WMMA per 6 fragment loads.
__global__ void k_gemm16(const _Float16* __restrict__ A, const _Float16* __restrict__ W,
                         const float* __restrict__ bias, float* __restrict__ D,
                         int Kp, int O) {
  int m0 = blockIdx.x * 64, o0 = blockIdx.y * 32;
  v8f acc[4][2];
#pragma unroll
  for (int i = 0; i < 4; ++i)
#pragma unroll
    for (int j = 0; j < 2; ++j) acc[i][j] = vzero8();
  for (int k0 = 0; k0 < Kp; k0 += 32) {
    if (k0 + 32 < Kp) {  // -> global_prefetch_b8, pull next k-step toward WGP
      __builtin_prefetch(A + (size_t)(m0 + (threadIdx.x & 15)) * Kp + k0 + 32, 0, 1);
      __builtin_prefetch(W + (size_t)(o0 + (threadIdx.x & 15)) * Kp + k0 + 32, 0, 1);
    }
    v16h b0 = load_bT_f16(W, o0,      Kp, k0);
    v16h b1 = load_bT_f16(W, o0 + 16, Kp, k0);
#pragma unroll
    for (int i = 0; i < 4; ++i) {
      v16h a = load_a_f16(A, m0 + 16 * i, Kp, k0);
      acc[i][0] = wmma_f16(a, b0, acc[i][0]);
      acc[i][1] = wmma_f16(a, b1, acc[i][1]);
    }
  }
  int lane = threadIdx.x & 31, col = lane & 15, half = lane >> 4;
#pragma unroll
  for (int j = 0; j < 2; ++j) {
    float bv = bias ? bias[o0 + 16 * j + col] : 0.f;
#pragma unroll
    for (int i = 0; i < 4; ++i)
#pragma unroll
      for (int q = 0; q < 8; ++q)
        D[(size_t)(m0 + 16 * i + q + 8 * half) * O + o0 + 16 * j + col] =
            acc[i][j][q] + bv;
  }
}

// ---------------- attention epilogue: h -> logits -> softmax over K ----------------
__global__ void k_att(const float* __restrict__ g1, const float* __restrict__ g2,
                      const float* __restrict__ dist, const float* __restrict__ w_att1,
                      const float* __restrict__ w_att2, float* __restrict__ att,
                      int a1cols, int rbfOff) {
  int wv = threadIdx.x >> 5, lane = threadIdx.x & 31;
  int bn = blockIdx.x * 8 + wv;
  __shared__ float srbf[8][16];
  float lg[8];
  const float* g1r = g1 + (size_t)bn * 64;
#pragma unroll
  for (int k = 0; k < Knn; ++k) {
    long long row = (long long)bn * Knn + k;
    float d = dist[row];
    if (lane < 16) {
      float t = d - (5.0f / 15.0f) * (float)lane;   // linspace(0,5,16)
      float e = __expf(-10.f * t * t);
      srbf[wv][lane] = fminf(fmaxf(e, 1e-10f), 1.f);
    }
    __syncthreads();
    float c = 0.f;
#pragma unroll
    for (int t = 0; t < 2; ++t) {
      int o = lane + 32 * t;
      float v = g1r[o] + g2[(size_t)row * 64 + o];
      const float* wr = w_att1 + (size_t)o * a1cols + rbfOff;
#pragma unroll
      for (int j = 0; j < 16; ++j) v += srbf[wv][j] * wr[j];
      v = leaky(v, 0.2f);
      c += v * w_att2[o];
    }
    __syncthreads();
#pragma unroll
    for (int off = 16; off; off >>= 1) c += __shfl_xor(c, off, 32);
    lg[k] = c;
  }
  float mx = lg[0];
#pragma unroll
  for (int k = 1; k < Knn; ++k) mx = fmaxf(mx, lg[k]);
  float e[8], s = 0.f;
#pragma unroll
  for (int k = 0; k < Knn; ++k) { e[k] = __expf(lg[k] - mx); s += e[k]; }
  if (lane < Knn) att[(size_t)bn * Knn + lane] = e[lane] / s;
}

// ------- upd pass1: per-M-tile BN column partial sums, 4Mx2O blocked (deterministic) -----
// grid: (BNK/64, cout/32)
__global__ void k_updpass1(const _Float16* __restrict__ neigh, const float* __restrict__ att,
                           const _Float16* __restrict__ Wu, float* __restrict__ psum,
                           float* __restrict__ psq, int Kp, int cout) {
  int m0 = blockIdx.x * 64, o0 = blockIdx.y * 32, lane = threadIdx.x & 31;
  _Float16 s[4];
#pragma unroll
  for (int i = 0; i < 4; ++i) s[i] = (_Float16)att[m0 + 16 * i + (lane & 15)];
  v8f acc[4][2];
#pragma unroll
  for (int i = 0; i < 4; ++i)
#pragma unroll
    for (int j = 0; j < 2; ++j) acc[i][j] = vzero8();
  for (int k0 = 0; k0 < Kp; k0 += 32) {
    v16h b0 = load_bT_f16(Wu, o0,      Kp, k0);
    v16h b1 = load_bT_f16(Wu, o0 + 16, Kp, k0);
#pragma unroll
    for (int i = 0; i < 4; ++i) {
      v16h a = load_a_f16(neigh, m0 + 16 * i, Kp, k0);
#pragma unroll
      for (int q = 0; q < 16; ++q) a[q] = a[q] * s[i];
      acc[i][0] = wmma_f16(a, b0, acc[i][0]);
      acc[i][1] = wmma_f16(a, b1, acc[i][1]);
    }
  }
#pragma unroll
  for (int i = 0; i < 4; ++i)
#pragma unroll
    for (int j = 0; j < 2; ++j) {
      float s1 = 0.f, s2 = 0.f;
#pragma unroll
      for (int q = 0; q < 8; ++q) { s1 += acc[i][j][q]; s2 += acc[i][j][q] * acc[i][j][q]; }
      s1 += __shfl_xor(s1, 16, 32);
      s2 += __shfl_xor(s2, 16, 32);
      if (lane < 16) {
        size_t mt = (size_t)(blockIdx.x * 4 + i);
        psum[mt * cout + o0 + 16 * j + lane] = s1;
        psq [mt * cout + o0 + 16 * j + lane] = s2;
      }
    }
}

// reduce partial columns: out[c] = sum_r P[r*cols+c]  (fixed order -> deterministic)
__global__ void k_redcol(const float* __restrict__ P, int rows, int cols,
                         float* __restrict__ out) {
  int c = blockIdx.x;
  float s = 0.f;
  for (int r = threadIdx.x; r < rows; r += blockDim.x) s += P[(size_t)r * cols + c];
  __shared__ float buf[256];
  buf[threadIdx.x] = s; __syncthreads();
  for (int off = 128; off; off >>= 1) {
    if (threadIdx.x < off) buf[threadIdx.x] += buf[threadIdx.x + off];
    __syncthreads();
  }
  if (threadIdx.x == 0) out[c] = buf[0];
}

__global__ void k_bnfin(const float* __restrict__ ssum, const float* __restrict__ ssq,
                        const float* __restrict__ g, const float* __restrict__ bb,
                        float* __restrict__ scale, float* __restrict__ shift,
                        int C, float invCount) {
  int c = blockIdx.x * blockDim.x + threadIdx.x;
  if (c >= C) return;
  float m = ssum[c] * invCount;
  float var = fmaxf(ssq[c] * invCount - m * m, 0.f);
  float sc = g[c] * rsqrtf(var + 1e-5f);
  scale[c] = sc; shift[c] = bb[c] - m * sc;
}

// ------- upd pass2: BN+leaky(upd)+0.1*res, mean over K, 2Mx2O blocked, dual GEMM --------
// grid: (BNK/32, cout/32)
__global__ void k_updpass2(const _Float16* __restrict__ neigh, const float* __restrict__ att,
                           const _Float16* __restrict__ Wu, const _Float16* __restrict__ Wr,
                           const float* __restrict__ scale, const float* __restrict__ shift,
                           float* __restrict__ feats, _Float16* __restrict__ xtout,
                           int Kp, int cout, int chanBase) {
  int m0 = blockIdx.x * 32, o0 = blockIdx.y * 32, lane = threadIdx.x & 31;
  int col = lane & 15, half = lane >> 4;
  _Float16 s[2];
#pragma unroll
  for (int i = 0; i < 2; ++i) s[i] = (_Float16)att[m0 + 16 * i + col];
  v8f au[2][2], ar[2][2];
#pragma unroll
  for (int i = 0; i < 2; ++i)
#pragma unroll
    for (int j = 0; j < 2; ++j) { au[i][j] = vzero8(); ar[i][j] = vzero8(); }
  for (int k0 = 0; k0 < Kp; k0 += 32) {
    v16h bu0 = load_bT_f16(Wu, o0,      Kp, k0);
    v16h bu1 = load_bT_f16(Wu, o0 + 16, Kp, k0);
    v16h br0 = load_bT_f16(Wr, o0,      Kp, k0);
    v16h br1 = load_bT_f16(Wr, o0 + 16, Kp, k0);
#pragma unroll
    for (int i = 0; i < 2; ++i) {
      v16h a = load_a_f16(neigh, m0 + 16 * i, Kp, k0);
      v16h as = a;
#pragma unroll
      for (int q = 0; q < 16; ++q) as[q] = as[q] * s[i];
      au[i][0] = wmma_f16(as, bu0, au[i][0]);
      au[i][1] = wmma_f16(as, bu1, au[i][1]);
      ar[i][0] = wmma_f16(a,  br0, ar[i][0]);
      ar[i][1] = wmma_f16(a,  br1, ar[i][1]);
    }
  }
#pragma unroll
  for (int j = 0; j < 2; ++j) {
    int o = o0 + 16 * j + col;
    float sc = scale[o], sh = shift[o];
#pragma unroll
    for (int i = 0; i < 2; ++i) {
      float fs = 0.f;
#pragma unroll
      for (int q = 0; q < 8; ++q) {   // 8 D-VGPRs = 8 rows = one (b,n) group (K=8)
        float u = leaky(sc * au[i][j][q] + sh, 0.02f);
        fs += u + 0.1f * ar[i][j][q];
      }
      float feat = fs * 0.125f;
      int grp = ((m0 + 16 * i) >> 3) + half;   // = b*N + n
      int b = grp / Npts, n = grp % Npts;
      feats[((size_t)b * EMB + chanBase + o) * Npts + n] = feat;
      xtout[(size_t)grp * cout + o] = (_Float16)feat;
    }
  }
}

// ---------------- final head ----------------
__global__ void k_transpose(const float* __restrict__ feats, _Float16* __restrict__ xc16) {
  long long total = (long long)BN * EMB;
  for (long long i = (long long)blockIdx.x * blockDim.x + threadIdx.x; i < total;
       i += (long long)gridDim.x * blockDim.x) {
    int c = (int)(i % EMB); long long bn = i / EMB;
    int b = (int)(bn / Npts), n = (int)(bn % Npts);
    xc16[i] = (_Float16)feats[((size_t)b * EMB + c) * Npts + n];
  }
}

// y (rows x cols) -> column sum / sumsq (per channel), deterministic
__global__ void k_colstats(const float* __restrict__ X, int rows, int cols,
                           float* __restrict__ ssum, float* __restrict__ ssq) {
  int c = blockIdx.x;
  float s1 = 0.f, s2 = 0.f;
  for (int r = threadIdx.x; r < rows; r += blockDim.x) {
    float v = X[(size_t)r * cols + c];
    s1 += v; s2 += v * v;
  }
  __shared__ float b1[256], b2[256];
  b1[threadIdx.x] = s1; b2[threadIdx.x] = s2; __syncthreads();
  for (int off = 128; off; off >>= 1) {
    if (threadIdx.x < off) { b1[threadIdx.x] += b1[threadIdx.x + off];
                             b2[threadIdx.x] += b2[threadIdx.x + off]; }
    __syncthreads();
  }
  if (threadIdx.x == 0) { ssum[c] = b1[0]; ssq[c] = b2[0]; }
}

// per (b,o): max and mean over N of leaky(bn(y)) -> pool (B,1920)
__global__ void k_pool(const float* __restrict__ y, const float* __restrict__ scale,
                       const float* __restrict__ shift, float* __restrict__ pool) {
  int wid = (blockIdx.x * blockDim.x + threadIdx.x) >> 5;  // b*960+o
  int lane = threadIdx.x & 31;
  int b = wid / EMB, o = wid % EMB;
  float sc = scale[o], sh = shift[o];
  float mx = -3.4e38f, sm = 0.f;
  for (int n = lane; n < Npts; n += 32) {
    float v = leaky(sc * y[((size_t)b * Npts + n) * EMB + o] + sh, 0.2f);
    mx = fmaxf(mx, v); sm += v;
  }
#pragma unroll
  for (int off = 16; off; off >>= 1) {
    mx = fmaxf(mx, __shfl_xor(mx, off, 32));
    sm += __shfl_xor(sm, off, 32);
  }
  if (lane == 0) {
    pool[(size_t)b * (2 * EMB) + o]        = mx;
    pool[(size_t)b * (2 * EMB) + EMB + o]  = sm * (1.f / Npts);
  }
}

__global__ void k_out(const float* __restrict__ mpre, const float* __restrict__ scale,
                      const float* __restrict__ shift, float* __restrict__ out) {
  int i = blockIdx.x * blockDim.x + threadIdx.x;
  if (i >= Bsz * EMB) return;
  int o = i % EMB;
  out[i] = leaky(scale[o] * mpre[i] + shift[o], 0.2f);
}

// ---------------- host ----------------
extern "C" void kernel_launch(void* const* d_in, const int* in_sizes, int n_in,
                              void* d_out, int out_size, void* d_ws, size_t ws_size,
                              hipStream_t stream) {
  const float* x = (const float*)d_in[0];
  const float *w_att1[4], *w_att2[4], *w_upd[4], *bn_g[4], *bn_b[4], *w_res[4];
  for (int i = 0; i < 4; ++i) {
    w_att1[i] = (const float*)d_in[2 + 6 * i + 0];
    w_att2[i] = (const float*)d_in[2 + 6 * i + 1];
    w_upd[i]  = (const float*)d_in[2 + 6 * i + 2];
    bn_g[i]   = (const float*)d_in[2 + 6 * i + 3];
    bn_b[i]   = (const float*)d_in[2 + 6 * i + 4];
    w_res[i]  = (const float*)d_in[2 + 6 * i + 5];
  }
  const float* conv_w     = (const float*)d_in[26];
  const float* conv_bn_g  = (const float*)d_in[27];
  const float* conv_bn_b  = (const float*)d_in[28];
  const float* merge_w    = (const float*)d_in[29];
  const float* merge_b    = (const float*)d_in[30];
  const float* merge_bn_g = (const float*)d_in[31];
  const float* merge_bn_b = (const float*)d_in[32];

  // ---- workspace carve-up (256B aligned regions) ----
  char* Wb = (char*)d_ws;
  size_t off = 0;
  auto carve = [&](size_t bytes) -> char* {
    char* p = Wb + off;
    off = (off + bytes + 255) & ~(size_t)255;
    return p;
  };
  _Float16* xtA    = (_Float16*)carve((size_t)BN * 512 * 2);
  _Float16* xtB    = (_Float16*)carve((size_t)BN * 512 * 2);
  _Float16* cent16 = (_Float16*)carve((size_t)BN * 32 * 2);
  float*    xx     = (float*)   carve((size_t)BN * 4);
  float*    pair   = (float*)   carve((size_t)BN * Npts * 4);
  int*      idx    = (int*)     carve((size_t)BN * Knn * 4);
  float*    dist   = (float*)   carve((size_t)BN * Knn * 4);
  _Float16* neigh  = (_Float16*)carve((size_t)BNK * 256 * 2);   // reused as ybuf
  float*    g1     = (float*)   carve((size_t)BN * 64 * 4);
  float*    g2     = (float*)   carve((size_t)BNK * 64 * 4);    // reused: partials / xc16
  float*    att    = (float*)   carve((size_t)BNK * 4);
  float*    ssum   = (float*)   carve(1024 * 4);
  float*    ssq    = (float*)   carve(1024 * 4);
  float*    scale  = (float*)   carve(1024 * 4);
  float*    shift  = (float*)   carve(1024 * 4);
  float*    feats  = (float*)   carve((size_t)Bsz * EMB * Npts * 4);
  float*    pool   = (float*)   carve((size_t)Bsz * 2 * EMB * 4);
  _Float16* pool16 = (_Float16*)carve((size_t)Bsz * 2 * EMB * 2);
  float*    mpre   = (float*)   carve((size_t)Bsz * EMB * 4);
  _Float16* w16ac  = (_Float16*)carve(64 * 256 * 2);
  _Float16* w16an  = (_Float16*)carve(64 * 256 * 2);
  _Float16* w16u   = (_Float16*)carve(512 * 256 * 2);
  _Float16* w16r   = (_Float16*)carve(512 * 256 * 2);
  _Float16* w16c   = (_Float16*)carve((size_t)960 * 960 * 2);
  _Float16* w16m   = (_Float16*)carve((size_t)960 * 1920 * 2);
  (void)ws_size; (void)in_sizes; (void)n_in; (void)out_size;

  float*    ybuf = (float*)neigh;                 // head-stage reuse
  _Float16* xc16 = (_Float16*)g2;
  float*    psum = g2;                            // pass-1 partials (per-layer reuse)
  float*    psq  = g2 + (size_t)(BNK / 16) * 512; // worst case layer: fits g2 exactly

  const int Cp[4]    = {32, 64, 128, 256};
  const int eff[4]   = {18, 64, 128, 256};
  const int effp[4]  = {32, 64, 128, 256};
  const int cout[4]  = {64, 128, 256, 512};
  const int a1c[4]   = {52, 144, 272, 528};
  const int chOff[4] = {3, 0, 0, 0};
  const int chBase[4]= {0, 64, 192, 448};

  k_prep0<<<(BN + 255) / 256, 256, 0, stream>>>(x, xtA);

  _Float16* xtcur = xtA;
  _Float16* xtnxt = xtB;

  for (int L = 0; L < 4; ++L) {
    int cp = Cp[L], ef = eff[L], efp = effp[L], co = cout[L];

    k_rowsq<<<(BN + 255) / 256, 256, 0, stream>>>(xtcur, cp, xx);
    k_gram<<<dim3(Npts / 32, Npts / 32, Bsz), 32, 0, stream>>>(xtcur, xx, pair, cp);
    k_topk<<<BN / 8, 256, 0, stream>>>(pair, idx, dist);

    // f16 weight tiles (K-padded to efp)
    long long wtot = (long long)64 * efp;
    k_cvt_w<<<(unsigned)((wtot + 255) / 256), 256, 0, stream>>>(w_att1[L], w16ac, 64, ef, a1c[L], 0,  efp);
    k_cvt_w<<<(unsigned)((wtot + 255) / 256), 256, 0, stream>>>(w_att1[L], w16an, 64, ef, a1c[L], ef, efp);
    long long utot = (long long)co * efp;
    k_cvt_w<<<(unsigned)((utot + 255) / 256), 256, 0, stream>>>(w_upd[L], w16u, co, ef, ef, 0, efp);
    k_cvt_w<<<(unsigned)((utot + 255) / 256), 256, 0, stream>>>(w_res[L], w16r, co, ef, ef, 0, efp);

    const _Float16* centp = xtcur;
    if (L == 0) {
      long long ct = (long long)BN * efp;
      k_centcopy<<<(unsigned)((ct + 255) / 256), 256, 0, stream>>>(xtcur, cent16, cp, chOff[L], ef, efp);
      centp = cent16;
    }

    long long ntot = (long long)BNK * efp;
    k_gather<<<(unsigned)((ntot + 255) / 256), 256, 0, stream>>>(xtcur, idx, neigh, cp, ef, efp, chOff[L], ntot);

    k_gemm16<<<dim3(BN / 64, 64 / 32), 32, 0, stream>>>(centp, w16ac, nullptr, g1, efp, 64);
    k_gemm16<<<dim3(BNK / 64, 64 / 32), 32, 0, stream>>>(neigh, w16an, nullptr, g2, efp, 64);
    k_att<<<BN / 8, 256, 0, stream>>>(g1, g2, dist, w_att1[L], w_att2[L], att, a1c[L], 2 * ef);

    // BN stats: deterministic two-stage (tile partials in g2 region, then tree-reduce)
    k_updpass1<<<dim3(BNK / 64, co / 32), 32, 0, stream>>>(neigh, att, w16u, psum, psq, efp, co);
    k_redcol<<<co, 256, 0, stream>>>(psum, BNK / 16, co, ssum);
    k_redcol<<<co, 256, 0, stream>>>(psq,  BNK / 16, co, ssq);
    k_bnfin<<<(co + 255) / 256, 256, 0, stream>>>(ssum, ssq, bn_g[L], bn_b[L], scale, shift, co, 1.f / (float)BNK);

    k_updpass2<<<dim3(BNK / 32, co / 32), 32, 0, stream>>>(neigh, att, w16u, w16r, scale, shift,
                                                           feats, xtnxt, efp, co, chBase[L]);
    _Float16* t = xtcur; xtcur = xtnxt; xtnxt = t;
  }

  // ---- head: conv (960x960) + BN + pools + merge + BN ----
  long long ttot = (long long)BN * EMB;
  k_transpose<<<(unsigned)((ttot + 255) / 256), 256, 0, stream>>>(feats, xc16);
  long long ctot = (long long)960 * 960;
  k_cvt_w<<<(unsigned)((ctot + 255) / 256), 256, 0, stream>>>(conv_w, w16c, 960, 960, 960, 0, 960);
  k_gemm16<<<dim3(BN / 64, EMB / 32), 32, 0, stream>>>(xc16, w16c, nullptr, ybuf, 960, EMB);
  k_colstats<<<EMB, 256, 0, stream>>>(ybuf, BN, EMB, ssum, ssq);
  k_bnfin<<<(EMB + 255) / 256, 256, 0, stream>>>(ssum, ssq, conv_bn_g, conv_bn_b, scale, shift, EMB, 1.f / (float)BN);
  k_pool<<<(Bsz * EMB) / 8, 256, 0, stream>>>(ybuf, scale, shift, pool);

  long long ptot = (long long)Bsz * 1920;
  k_cvt_w<<<(unsigned)((ptot + 255) / 256), 256, 0, stream>>>(pool, pool16, Bsz, 1920, 1920, 0, 1920);
  long long mtot = (long long)960 * 1920;
  k_cvt_w<<<(unsigned)((mtot + 255) / 256), 256, 0, stream>>>(merge_w, w16m, 960, 1920, 1920, 0, 1920);
  k_gemm16<<<dim3(Bsz / 64, EMB / 32), 32, 0, stream>>>(pool16, w16m, merge_b, mpre, 1920, EMB);
  k_colstats<<<EMB, 256, 0, stream>>>(mpre, Bsz, EMB, ssum, ssq);
  k_bnfin<<<(EMB + 255) / 256, 256, 0, stream>>>(ssum, ssq, merge_bn_g, merge_bn_b, scale, shift, EMB, 1.f / (float)Bsz);
  k_out<<<(Bsz * EMB + 255) / 256, 256, 0, stream>>>(mpre, scale, shift, (float*)d_out);
}
